// BailingMoeV2_5LinearAttention_89721866814157
// MI455X (gfx1250) — matrix operand order repaired and checked by reading, not compiled, not run on hardware
//
#include <hip/hip_runtime.h>
#include <hip/hip_bf16.h>
#include <math.h>

typedef __attribute__((ext_vector_type(16))) __bf16 v16bf;
typedef __attribute__((ext_vector_type(8)))  float  v8f;
typedef __attribute__((ext_vector_type(2)))  float  v2f;
typedef __attribute__((ext_vector_type(4)))  unsigned int v4u;
typedef __attribute__((ext_vector_type(8)))  int    v8i;
typedef __attribute__((ext_vector_type(4)))  int    v4i;

namespace cfg {
constexpr int Bn = 4, Sn = 4096, HIDn = 2048, Hn = 16, Dn = 128;
constexpr int Tn = Bn * Sn;              // 16384 tokens
constexpr int QKVW = 3 * Hn * Dn;        // 6144
constexpr int CHn = 64, NCn = Sn / CHn;  // chunk 64, 64 chunks
constexpr float EPSc = 1e-6f;
constexpr float SCALEc = 0.08838834764831845f;  // 128^-0.5
}

struct GammaVec { float g[16]; };

// ---------------------------------------------------------------------------
// f32 -> bf16 cast (grid-stride)
// ---------------------------------------------------------------------------
__global__ __launch_bounds__(256)
void cast_f32_bf16(const float* __restrict__ x, __bf16* __restrict__ y, long long n) {
  for (long long i = (long long)blockIdx.x * 256 + threadIdx.x; i < n;
       i += (long long)gridDim.x * 256)
    y[i] = (__bf16)x[i];
}

// ---------------------------------------------------------------------------
// W [K,N] f32  ->  Wt [N,K] bf16  (32x32 LDS tile transpose)
// ---------------------------------------------------------------------------
__global__ __launch_bounds__(256)
void transpose_cast(const float* __restrict__ W, __bf16* __restrict__ Wt,
                    int K, int N) {
  __shared__ float tile[32][33];
  int n0 = blockIdx.x * 32, k0 = blockIdx.y * 32;
  int tx = threadIdx.x & 31, ty = threadIdx.x >> 5;  // 32 x 8
  #pragma unroll
  for (int r = 0; r < 32; r += 8)
    tile[ty + r][tx] = W[(size_t)(k0 + ty + r) * N + (n0 + tx)];
  __syncthreads();
  #pragma unroll
  for (int r = 0; r < 32; r += 8)
    Wt[(size_t)(n0 + ty + r) * K + (k0 + tx)] = (__bf16)tile[tx][ty + r];
}

// ---------------------------------------------------------------------------
// C[M,N] f32 = A[M,K]bf16 @ Bt[N,K]bf16   (Bt is transposed weights)
// block: 256 threads = 8 waves stacked in M; tile 128(M) x 64(N); wave: 16x64.
// EPI: 0 = plain store, 1 = sigmoid store (for gating branch)
// ---------------------------------------------------------------------------
template <int EPI>
__global__ __launch_bounds__(256)
void gemm_bf16(const __bf16* __restrict__ A, const __bf16* __restrict__ Bt,
               float* __restrict__ C, int M, int N, int K) {
  const int wave = threadIdx.x >> 5;
  const int lane = threadIdx.x & 31;
  const int l16  = lane & 15;
  const int hi   = lane >> 4;                 // 0 | 1
  const int mBase = blockIdx.x * 128 + wave * 16;
  const int nBase = blockIdx.y * 64;
  const int kbA = hi * 8;                     // A lane K-base (16-bit A layout)
  const int kbB = hi * 8;                     // B lane K-base
  const __bf16* aPtr = A + (size_t)(mBase + l16) * K;

  v8f acc[4] = {v8f{}, v8f{}, v8f{}, v8f{}};

  for (int kk = 0; kk < K; kk += 32) {
    __builtin_prefetch(aPtr + kk + 256, 0, 1);   // global_prefetch_b8
    union { v16bf v; uint4 u[2]; } a;
    a.u[0] = *(const uint4*)(aPtr + kk + kbA);        // K = kbA .. kbA+7
    a.u[1] = *(const uint4*)(aPtr + kk + kbA + 16);   // K = kbA+16 .. kbA+23
    #pragma unroll
    for (int nt = 0; nt < 4; ++nt) {
      const __bf16* bPtr = Bt + (size_t)(nBase + nt * 16 + l16) * K + kk + kbB;
      union { v16bf v; uint4 u[2]; } b;
      b.u[0] = *(const uint4*)(bPtr);                 // K = kbB .. kbB+7
      b.u[1] = *(const uint4*)(bPtr + 16);            // K = kbB+16 .. kbB+23
      acc[nt] = __builtin_amdgcn_wmma_f32_16x16x32_bf16(
          false, a.v, false, b.v, (short)0, acc[nt], false, false);
    }
  }

  #pragma unroll
  for (int nt = 0; nt < 4; ++nt) {
    #pragma unroll
    for (int r = 0; r < 8; ++r) {
      int row = mBase + r + hi * 8;
      int col = nBase + nt * 16 + l16;
      float v = acc[nt][r];
      if (EPI == 1) v = 1.0f / (1.0f + __expf(-v));
      C[(size_t)row * N + col] = v;
    }
  }
}

// ---------------------------------------------------------------------------
// per-(token,head) RMSNorm + RoPE on q and k, in place in qkv[T, 3*H*D].
// one wave per (token, head, which). lane holds dims {l, l+32, l+64, l+96};
// RoPE pair (d, d+32) for d<32 is in-lane.
// ---------------------------------------------------------------------------
__global__ __launch_bounds__(256)
void qk_norm_rope(float* __restrict__ qkv, const int* __restrict__ pos,
                  const float* __restrict__ qw, const float* __restrict__ kw) {
  using namespace cfg;
  int wid   = blockIdx.x * 8 + (threadIdx.x >> 5);
  int lane  = threadIdx.x & 31;
  int which = wid & 1;
  int h     = (wid >> 1) & 15;
  int t     = wid >> 5;
  float* p = qkv + (size_t)t * QKVW + which * (Hn * Dn) + h * Dn;

  float x0 = p[lane], x1 = p[lane + 32], x2 = p[lane + 64], x3 = p[lane + 96];
  float ss = x0 * x0 + x1 * x1 + x2 * x2 + x3 * x3;
  #pragma unroll
  for (int m = 16; m; m >>= 1) ss += __shfl_xor(ss, m, 32);
  float inv = rsqrtf(ss * (1.0f / 128.0f) + EPSc);
  const float* w = which ? kw : qw;
  x0 *= inv * w[lane];      x1 *= inv * w[lane + 32];
  x2 *= inv * w[lane + 64]; x3 *= inv * w[lane + 96];

  // RoPE: inv_freq[d] = 10000^(-d/32), d = lane (0..31)
  float ang = (float)pos[t] * __expf(-(float)lane * 0.28782313662427653f);
  float s, c;
  __sincosf(ang, &s, &c);
  float n0 = x0 * c - x1 * s;
  float n1 = x1 * c + x0 * s;
  p[lane] = n0; p[lane + 32] = n1; p[lane + 64] = x2; p[lane + 96] = x3;
}

// ---------------------------------------------------------------------------
// TDM helper: async DMA of a rows x 128-f32 tile (row stride in elements)
// into LDS with +4-DWORD padding after every 128 DWORDs (-> pitch 132 floats).
// D# packing per CDNA5 ISA (2D tile, groups 2/3=0).
// This toolchain exposes the 6-arg builtin form; the extra int32x8 carries the
// remaining (unused) descriptor state and is zero-filled for a 2D tile.
// ---------------------------------------------------------------------------
__device__ __forceinline__ void tdm_load_2d(const float* gptr, unsigned lds_off,
                                            int rows, int row_stride_elems) {
  unsigned long long ga = (unsigned long long)(uintptr_t)gptr;
  v4u g0;
  g0[0] = 1u;                                   // count=1, user descriptor
  g0[1] = lds_off;                              // lds_addr (bytes)
  g0[2] = (unsigned)ga;                         // global_addr[31:0]
  g0[3] = (unsigned)(ga >> 32) | (2u << 30);    // global_addr[56:32], type=2
  v8i g1;
  g1[0] = (int)((2u << 16)      // data_size = 4 bytes
              | (1u << 20)      // pad_enable
              | (6u << 22)      // pad_interval: 128 DWORDs
              | (3u << 25));    // pad_amount: 4 DWORDs
  g1[1] = (int)(128u << 16);                    // tensor_dim0 = 128 (lo16)
  g1[2] = (int)((unsigned)rows << 16);          // tensor_dim0 hi=0 | tensor_dim1 lo
  g1[3] = (int)(128u << 16);                    // tensor_dim1 hi=0 | tile_dim0=128
  g1[4] = (int)(unsigned)rows;                  // tile_dim1 = rows | tile_dim2 = 0
  g1[5] = row_stride_elems;                     // tensor_dim0_stride[31:0]
  g1[6] = 0;                                    // stride hi | dim1_stride lo
  g1[7] = 0;                                    // dim1_stride hi
  v4i zero4 = {0, 0, 0, 0};
  v8i zero8 = {0, 0, 0, 0, 0, 0, 0, 0};
  __builtin_amdgcn_tensor_load_to_lds(g0, g1, zero4, zero4, zero8, 0);
}

__device__ __forceinline__ unsigned lds_addr_of(const void* p) {
  // generic pointer to LDS: low 32 bits are the wave-relative LDS byte address
  return (unsigned)(uintptr_t)p;
}

// ---------------------------------------------------------------------------
// GLA chunked scan, f32 WMMA (v_wmma_f32_16x16x4_f32) + TDM double buffering.
// one block (16 waves / 512 threads) per (b,h); state h[128][128] in LDS.
// ---------------------------------------------------------------------------
#define HP 132   // h   row pitch (floats)
#define QP 132   // q/k/v row pitch
#define SP 68    // scores row pitch

__global__ __launch_bounds__(512)
void gla_kernel(const float* __restrict__ qkv, const float* __restrict__ h0,
                float* __restrict__ og, float* __restrict__ hT, GammaVec gv) {
  using namespace cfg;
  __shared__ float hS[128 * HP];
  __shared__ float qS[2][64 * QP];
  __shared__ float kS[2][64 * QP];
  __shared__ float vS[2][64 * QP];
  __shared__ float sS[64 * SP];

  const int bh = blockIdx.x;            // 0..63
  const int b = bh >> 4, h = bh & 15;
  const float gamma = gv.g[h];
  const float dchunk = __expf(gamma * 64.0f);
  const int wave = threadIdx.x >> 5, lane = threadIdx.x & 31;
  const int l16 = lane & 15, hiL = lane >> 4;
  const int kb = hiL * 2;               // f32 WMMA lane K-base

  // prologue: TDM-load initial state + chunk 0 tiles (wave 0 issues for all)
  if (wave == 0) {
    tdm_load_2d(h0 + (size_t)bh * Dn * Dn, lds_addr_of(hS), 128, 128);
    const float* gp = qkv + ((size_t)b * Sn) * QKVW + h * Dn;
    tdm_load_2d(gp,              lds_addr_of(qS[0]), 64, QKVW);
    tdm_load_2d(gp + Hn * Dn,    lds_addr_of(kS[0]), 64, QKVW);
    tdm_load_2d(gp + 2 * Hn * Dn,lds_addr_of(vS[0]), 64, QKVW);
  }

  for (int ch = 0; ch < NCn; ++ch) {
    const int cur = ch & 1;
    // ---- prefetch next chunk into the other buffer, wait for current tiles
    if (wave == 0) {
      if (ch + 1 < NCn) {
        const float* gp =
            qkv + ((size_t)b * Sn + (size_t)(ch + 1) * CHn) * QKVW + h * Dn;
        tdm_load_2d(gp,               lds_addr_of(qS[cur ^ 1]), 64, QKVW);
        tdm_load_2d(gp + Hn * Dn,     lds_addr_of(kS[cur ^ 1]), 64, QKVW);
        tdm_load_2d(gp + 2 * Hn * Dn, lds_addr_of(vS[cur ^ 1]), 64, QKVW);
        __builtin_amdgcn_s_wait_tensorcnt(3);  // in-order: current chunk done
      } else {
        __builtin_amdgcn_s_wait_tensorcnt(0);
      }
    }
    __syncthreads();   // current tiles (and, first iter, hS) visible to all

    const float* qC = qS[cur];
    const float* kC = kS[cur];
    const float* vC = vS[cur];
    const size_t tok0 = (size_t)b * Sn + (size_t)ch * CHn;

    // ---- scores[64,64] = q @ k^T ; wave -> tile (mi, mj)
    {
      int mi = wave >> 2, mj = wave & 3;
      v8f sacc = {};
      const float* aBase = &qC[(mi * 16 + l16) * QP];
      const float* bBase = &kC[(mj * 16 + l16) * QP];
      for (int kk = 0; kk < 128; kk += 4) {
        v2f av = {aBase[kk + kb], aBase[kk + kb + 1]};
        v2f bv = {bBase[kk + kb], bBase[kk + kb + 1]};
        sacc = __builtin_amdgcn_wmma_f32_16x16x4_f32(
            false, av, false, bv, (short)0, sacc, false, false);
      }
      // causal decay mask, write to sS
      #pragma unroll
      for (int r = 0; r < 8; ++r) {
        int i = mi * 16 + r + hiL * 8;
        int j = mj * 16 + l16;
        float f = (i >= j) ? __expf(gamma * (float)(i - j)) : 0.0f;
        sS[i * SP + j] = sacc[r] * f;
      }
    }

    // ---- state-update accumulators: (k*dk)^T @ v ; wave -> 4 tiles of 8x8
    v8f hacc[4] = {v8f{}, v8f{}, v8f{}, v8f{}};
    for (int kk = 0; kk < 64; kk += 4) {
      int j0 = kk + kb, j1 = j0 + 1;
      float dk0 = __expf(gamma * (float)(63 - j0));
      float dk1 = __expf(gamma * (float)(63 - j1));
      #pragma unroll
      for (int q = 0; q < 4; ++q) {
        int ti = wave * 4 + q, di = ti >> 3, ei = ti & 7;
        v2f av = {kC[j0 * QP + di * 16 + l16] * dk0,
                  kC[j1 * QP + di * 16 + l16] * dk1};
        v2f bv = {vC[j0 * QP + ei * 16 + l16],
                  vC[j1 * QP + ei * 16 + l16]};
        hacc[q] = __builtin_amdgcn_wmma_f32_16x16x4_f32(
            false, av, false, bv, (short)0, hacc[q], false, false);
      }
    }
    __syncthreads();   // scores visible; all waves still reading OLD hS below

    // ---- o = dq*(q @ h) + s @ v ; wave -> 2 tiles of the 4x8 grid
    #pragma unroll
    for (int q = 0; q < 2; ++q) {
      int ti = wave * 2 + q;          // 0..31
      int mi2 = ti >> 3, ne = ti & 7;
      v8f acc = {};
      const float* aBase = &qC[(mi2 * 16 + l16) * QP];
      for (int kk = 0; kk < 128; kk += 4) {
        v2f av = {aBase[kk + kb], aBase[kk + kb + 1]};
        v2f bv = {hS[(kk + kb) * HP + ne * 16 + l16],
                  hS[(kk + kb + 1) * HP + ne * 16 + l16]};
        acc = __builtin_amdgcn_wmma_f32_16x16x4_f32(
            false, av, false, bv, (short)0, acc, false, false);
      }
      #pragma unroll
      for (int r = 0; r < 8; ++r)     // dq[i] = exp(gamma*(i+1)), inter only
        acc[r] *= __expf(gamma * (float)(mi2 * 16 + r + hiL * 8 + 1));
      const float* sBase = &sS[(mi2 * 16 + l16) * SP];
      for (int kk = 0; kk < 64; kk += 4) {
        v2f av = {sBase[kk + kb], sBase[kk + kb + 1]};
        v2f bv = {vC[(kk + kb) * QP + ne * 16 + l16],
                  vC[(kk + kb + 1) * QP + ne * 16 + l16]};
        acc = __builtin_amdgcn_wmma_f32_16x16x4_f32(
            false, av, false, bv, (short)0, acc, false, false);
      }
      #pragma unroll
      for (int r = 0; r < 8; ++r) {
        int i = mi2 * 16 + r + hiL * 8;
        og[(tok0 + i) * (size_t)HIDn + h * Dn + ne * 16 + l16] = acc[r] * SCALEc;
      }
    }
    __syncthreads();   // every wave finished reading old hS

    // ---- h <- dchunk*h + hacc  (each wave owns its 4 tiles exclusively)
    #pragma unroll
    for (int q = 0; q < 4; ++q) {
      int ti = wave * 4 + q, di = ti >> 3, ei = ti & 7;
      #pragma unroll
      for (int r = 0; r < 8; ++r) {
        int d = di * 16 + r + hiL * 8;
        int e = ei * 16 + l16;
        hS[d * HP + e] = dchunk * hS[d * HP + e] + hacc[q][r];
      }
    }
    __syncthreads();   // writeback visible; safe for next-chunk TDM overwrite
  }

  // final state out
  float* hout = hT + (size_t)bh * Dn * Dn;
  for (int i = threadIdx.x; i < 128 * 128; i += 512)
    hout[i] = hS[(i >> 7) * HP + (i & 127)];
}

// ---------------------------------------------------------------------------
// group RMS norm (8 groups of 256) * g_norm_w * sigmoid-gate -> bf16 for GEMM
// one block per (token, group); thread per element.
// ---------------------------------------------------------------------------
__global__ __launch_bounds__(256)
void gnorm_gate(const float* __restrict__ o, const float* __restrict__ gate_sig,
                const float* __restrict__ gw, __bf16* __restrict__ a2) {
  using namespace cfg;
  __shared__ float red[8];
  int t = blockIdx.x >> 3, g = blockIdx.x & 7;
  int idx = g * 256 + threadIdx.x;
  size_t off = (size_t)t * HIDn + idx;
  float v = o[off];
  float ss = v * v;
  #pragma unroll
  for (int m = 16; m; m >>= 1) ss += __shfl_xor(ss, m, 32);
  if ((threadIdx.x & 31) == 0) red[threadIdx.x >> 5] = ss;
  __syncthreads();
  float tot = 0.0f;
  #pragma unroll
  for (int i = 0; i < 8; ++i) tot += red[i];
  float inv = rsqrtf(tot * (1.0f / 256.0f) + EPSc);
  a2[off] = (__bf16)(v * inv * gw[idx] * gate_sig[off]);
}

// ---------------------------------------------------------------------------
// host launcher
// ---------------------------------------------------------------------------
extern "C" void kernel_launch(void* const* d_in, const int* in_sizes, int n_in,
                              void* d_out, int out_size, void* d_ws, size_t ws_size,
                              hipStream_t stream) {
  using namespace cfg;
  const int*   positions = (const int*)d_in[0];
  const float* hidden    = (const float*)d_in[1];
  const float* h0        = (const float*)d_in[2];
  const float* w_qkv     = (const float*)d_in[3];
  const float* w_g       = (const float*)d_in[4];
  const float* w_dense   = (const float*)d_in[5];
  const float* q_norm_w  = (const float*)d_in[6];
  const float* k_norm_w  = (const float*)d_in[7];
  const float* g_norm_w  = (const float*)d_in[8];
  (void)in_sizes; (void)n_in; (void)out_size; (void)ws_size;

  float* out = (float*)d_out;                     // [T, HID]
  float* hT  = out + (size_t)Tn * HIDn;           // [B, H, D, D]

  char* ws = (char*)d_ws;
  size_t off = 0;
  auto take = [&](size_t bytes) -> char* {
    char* p = ws + off;
    off += (bytes + 255) & ~(size_t)255;
    return p;
  };
  float*  qkv      = (float*) take((size_t)Tn * QKVW * 4);
  float*  gate_sig = (float*) take((size_t)Tn * HIDn * 4);
  float*  o_ws     = (float*) take((size_t)Tn * HIDn * 4);
  __bf16* hb       = (__bf16*)take((size_t)Tn * HIDn * 2);
  __bf16* a2       = (__bf16*)take((size_t)Tn * HIDn * 2);
  __bf16* wqkvT    = (__bf16*)take((size_t)QKVW * HIDn * 2);
  __bf16* wgT      = (__bf16*)take((size_t)HIDn * HIDn * 2);
  __bf16* wdT      = (__bf16*)take((size_t)HIDn * HIDn * 2);

  // ALiBi-derived per-head gamma (closed form: H=16 is a power of two)
  GammaVec gv;
  double decay = 1.0 - 7.0 / 31.0 + 1e-5;
  for (int i = 0; i < 16; ++i)
    gv.g[i] = (float)(-pow(2.0, -0.5 * (double)(i + 1)) * decay);

  cast_f32_bf16<<<4096, 256, 0, stream>>>(hidden, hb, (long long)Tn * HIDn);
  transpose_cast<<<dim3(QKVW / 32, HIDn / 32), 256, 0, stream>>>(w_qkv, wqkvT, HIDn, QKVW);
  transpose_cast<<<dim3(HIDn / 32, HIDn / 32), 256, 0, stream>>>(w_g, wgT, HIDn, HIDn);
  transpose_cast<<<dim3(HIDn / 32, HIDn / 32), 256, 0, stream>>>(w_dense, wdT, HIDn, HIDn);

  gemm_bf16<0><<<dim3(Tn / 128, QKVW / 64), 256, 0, stream>>>(hb, wqkvT, qkv, Tn, QKVW, HIDn);
  gemm_bf16<1><<<dim3(Tn / 128, HIDn / 64), 256, 0, stream>>>(hb, wgT, gate_sig, Tn, HIDn, HIDn);

  qk_norm_rope<<<(Tn * Hn * 2) / 8, 256, 0, stream>>>(qkv, positions, q_norm_w, k_norm_w);

  gla_kernel<<<Bn * Hn, 512, 0, stream>>>(qkv, h0, o_ws, hT, gv);

  gnorm_gate<<<Tn * 8, 256, 0, stream>>>(o_ws, gate_sig, g_norm_w, a2);

  gemm_bf16<0><<<dim3(Tn / 128, HIDn / 64), 256, 0, stream>>>(a2, wdT, out, Tn, HIDn, HIDn);
}
